// GCN_11914239279184
// MI455X (gfx1250) — matrix operand order repaired and checked by reading, not compiled, not run on hardware
//
#include <hip/hip_runtime.h>
#include <hip/hip_bf16.h>

#define N_NODES   50000
#define N_EDGES   800000
#define IN_DIM    128
#define HID_DIM   128
#define N_CLASSES 40

typedef __attribute__((ext_vector_type(16))) __bf16 v16bf;
typedef __attribute__((ext_vector_type(8)))  __bf16 v8bf;
typedef __attribute__((ext_vector_type(8)))  float  v8f;

static __device__ __forceinline__ void atomic_add_f32(float* p, float v) {
    // Non-returning, device-scope, relaxed -> native global_atomic_add_f32
    __hip_atomic_fetch_add(p, v, __ATOMIC_RELAXED, __HIP_MEMORY_SCOPE_AGENT);
}

// ---------------- degree / normalization ----------------

__global__ __launch_bounds__(256) void k_deg_init(float* deg) {
    int i = blockIdx.x * blockDim.x + threadIdx.x;
    if (i < N_NODES) deg[i] = 1.0f;   // self-loop contributes 1 to every node
}

__global__ __launch_bounds__(256) void k_deg_accum(const int* __restrict__ dst,
                                                   float* deg) {
    int e = blockIdx.x * blockDim.x + threadIdx.x;
    if (e < N_EDGES) atomic_add_f32(&deg[dst[e]], 1.0f);
}

__global__ __launch_bounds__(256) void k_dinv(float* deg) {
    int i = blockIdx.x * blockDim.x + threadIdx.x;
    if (i < N_NODES) deg[i] = rsqrtf(deg[i]);   // deg >= 1 always
}

// ---------------- weight pre-transpose + bf16 convert ----------------
// Wt[n * IN_DIM + k] = (bf16) W[k * out_dim + n]   (Wt is [out_dim x IN_DIM])
__global__ __launch_bounds__(256) void k_wt_bf16(const float* __restrict__ W,
                                                 __bf16* __restrict__ Wt,
                                                 int out_dim) {
    int idx = blockIdx.x * blockDim.x + threadIdx.x;
    if (idx >= out_dim * IN_DIM) return;
    int n = idx / IN_DIM;
    int k = idx % IN_DIM;
    Wt[idx] = (__bf16)W[(size_t)k * out_dim + n];
}

// ---------------- dense GEMM via bf16 WMMA, fused epilogue ----------------
// H[N x OUT]   = X[N x 128] * W[128 x OUT]            (raw product, for scatter gather)
// AGG[N x OUT] = dinv[row]^2 * H + bias[col]          (self-loop + bias init)
// One wave32 computes one 16x16 output tile; K loop in steps of 32.
template<int OUT>
__global__ __launch_bounds__(256) void k_gemm_bf16(const float*  __restrict__ X,
                                                   const __bf16* __restrict__ Wt,
                                                   const float*  __restrict__ dinv,
                                                   const float*  __restrict__ bias,
                                                   float* __restrict__ H,
                                                   float* __restrict__ AGG) {
    constexpr int TILES_N = (OUT + 15) / 16;
    constexpr int TILES_M = N_NODES / 16;       // 50000 is a multiple of 16
    const int wave = (blockIdx.x * blockDim.x + threadIdx.x) >> 5;
    const int lane = threadIdx.x & 31;
    const int tm = wave / TILES_N;
    const int tn = wave % TILES_N;
    if (tm >= TILES_M) return;

    const int m0   = tm * 16;
    const int n0   = tn * 16;
    const int half = lane >> 4;       // lane group 0..15 vs 16..31
    const int l    = lane & 15;
    const int arow = m0 + l;          // A row held by this lane
    const int bcol = n0 + l;          // B column held by this lane
    const bool colok = (bcol < OUT);

    v8f acc = {};
    #pragma unroll
    for (int k0 = 0; k0 < IN_DIM; k0 += 32) {
        // A fragment (16x32 bf16): lane group 0 holds K = k0+{0..7,16..23},
        // lane group 1 holds K = k0+{8..15,24..31}  (ISA 7.12.2)
        const float4* xa =
            (const float4*)(X + (size_t)arow * IN_DIM + k0 + half * 8);
        float4 a0 = xa[0], a1 = xa[1];      // K base+0..7
        float4 a2 = xa[4], a3 = xa[5];      // K base+16..23
        v16bf a;
        a[0]=(__bf16)a0.x; a[1]=(__bf16)a0.y; a[2]=(__bf16)a0.z; a[3]=(__bf16)a0.w;
        a[4]=(__bf16)a1.x; a[5]=(__bf16)a1.y; a[6]=(__bf16)a1.z; a[7]=(__bf16)a1.w;
        a[8]=(__bf16)a2.x; a[9]=(__bf16)a2.y; a[10]=(__bf16)a2.z; a[11]=(__bf16)a2.w;
        a[12]=(__bf16)a3.x; a[13]=(__bf16)a3.y; a[14]=(__bf16)a3.z; a[15]=(__bf16)a3.w;

        // B fragment (32x16 bf16): lane group 0 holds K = k0+{0..7,16..23},
        // lane group 1 holds K = k0+{8..15,24..31}; column = bcol.
        // Wt is [OUT x IN] row-major bf16 -> 2x b128 loads of 8 bf16.
        v16bf b;
        if (colok) {
            const v8bf* wr =
                (const v8bf*)(Wt + (size_t)bcol * IN_DIM + k0 + half * 8);
            v8bf b0 = wr[0], b1 = wr[2];
            #pragma unroll
            for (int i = 0; i < 8; ++i) { b[i] = b0[i]; b[8 + i] = b1[i]; }
        } else {
            #pragma unroll
            for (int i = 0; i < 16; ++i) b[i] = (__bf16)0.0f;
        }

        acc = __builtin_amdgcn_wmma_f32_16x16x32_bf16(
            /*neg_a=*/false, a, /*neg_b=*/false, b,
            /*c_mod=*/(short)0, acc, /*reuse_a=*/false, /*reuse_b=*/false);
    }

    // D layout: VGPR j -> row m0 + half*8 + j, col = n0 + (lane&15)
    if (colok) {
        float bv = bias[bcol];
        #pragma unroll
        for (int j = 0; j < 8; ++j) {
            int row = m0 + half * 8 + j;
            float v  = acc[j];
            float di = dinv[row];
            H  [(size_t)row * OUT + bcol] = v;
            AGG[(size_t)row * OUT + bcol] = di * di * v + bv;
        }
    }
}

// ---------------- edge scatter (L2-resident atomics) ----------------
// One wave per edge; 128 features -> float4 per lane, 4 native f32 atomics.
__global__ __launch_bounds__(256) void k_scatter128(const float* __restrict__ H,
                                                    const int* __restrict__ src,
                                                    const int* __restrict__ dst,
                                                    const float* __restrict__ dinv,
                                                    float* __restrict__ out) {
    long long t = (long long)blockIdx.x * blockDim.x + threadIdx.x;
    int e = (int)(t >> 5);
    if (e >= N_EDGES) return;
    int lane = threadIdx.x & 31;

    int s = src[e];
    int d = dst[e];
    float nrm = dinv[s] * dinv[d];

    // gfx1250 prefetch of an upcoming source row (global_prefetch_b8)
    if (e + 8 < N_EDGES)
        __builtin_prefetch(H + (size_t)src[e + 8] * HID_DIM, 0, 0);

    float4 v = ((const float4*)(H + (size_t)s * HID_DIM))[lane];
    float* o = out + (size_t)d * HID_DIM + lane * 4;
    atomic_add_f32(o + 0, nrm * v.x);
    atomic_add_f32(o + 1, nrm * v.y);
    atomic_add_f32(o + 2, nrm * v.z);
    atomic_add_f32(o + 3, nrm * v.w);
}

// 40 features -> lane f plus lanes 0..7 handle f+32
__global__ __launch_bounds__(256) void k_scatter40(const float* __restrict__ H,
                                                   const int* __restrict__ src,
                                                   const int* __restrict__ dst,
                                                   const float* __restrict__ dinv,
                                                   float* __restrict__ out) {
    long long t = (long long)blockIdx.x * blockDim.x + threadIdx.x;
    int e = (int)(t >> 5);
    if (e >= N_EDGES) return;
    int lane = threadIdx.x & 31;

    int s = src[e];
    int d = dst[e];
    float nrm = dinv[s] * dinv[d];

    const float* h = H + (size_t)s * N_CLASSES;
    float* o = out + (size_t)d * N_CLASSES;
    atomic_add_f32(o + lane, nrm * h[lane]);
    if (lane < 8) atomic_add_f32(o + 32 + lane, nrm * h[32 + lane]);
}

__global__ __launch_bounds__(256) void k_relu(float* a, long long n) {
    long long idx = (long long)blockIdx.x * blockDim.x + threadIdx.x;
    if (idx < n) a[idx] = fmaxf(a[idx], 0.0f);
}

// ---------------- host launch ----------------

extern "C" void kernel_launch(void* const* d_in, const int* in_sizes, int n_in,
                              void* d_out, int out_size, void* d_ws, size_t ws_size,
                              hipStream_t stream) {
    const float* x  = (const float*)d_in[0];
    const int*   ei = (const int*)d_in[1];
    const float* W1 = (const float*)d_in[2];
    const float* b1 = (const float*)d_in[3];
    const float* W2 = (const float*)d_in[4];
    const float* b2 = (const float*)d_in[5];
    float* out = (float*)d_out;

    const int* src = ei;             // edge_index[0]
    const int* dst = ei + N_EDGES;   // edge_index[1]

    // workspace layout
    float* deg  = (float*)d_ws;                          // N  (becomes dinv)
    float* hW1  = deg  + N_NODES;                        // N*128
    float* agg1 = hW1  + (size_t)N_NODES * HID_DIM;      // N*128
    float* hW2  = agg1 + (size_t)N_NODES * HID_DIM;      // N*40
    __bf16* W1t = (__bf16*)(hW2 + (size_t)N_NODES * N_CLASSES); // 128*128 bf16
    __bf16* W2t = W1t + (size_t)HID_DIM * IN_DIM;               // 40*128 bf16

    const int B = 256;
    auto blocks = [](long long n, int b) { return (int)((n + b - 1) / b); };

    // degrees -> dinv ; weight transpose+convert (independent, same stream)
    k_deg_init <<<blocks(N_NODES, B), B, 0, stream>>>(deg);
    k_deg_accum<<<blocks(N_EDGES, B), B, 0, stream>>>(dst, deg);
    k_dinv     <<<blocks(N_NODES, B), B, 0, stream>>>(deg);
    k_wt_bf16  <<<blocks(HID_DIM * IN_DIM, B), B, 0, stream>>>(W1, W1t, HID_DIM);
    k_wt_bf16  <<<blocks(N_CLASSES * IN_DIM, B), B, 0, stream>>>(W2, W2t, N_CLASSES);

    // layer 1: GEMM (+self-loop+bias epilogue) -> edge scatter -> ReLU
    {
        long long waves = (long long)(N_NODES / 16) * (HID_DIM / 16);
        k_gemm_bf16<HID_DIM><<<blocks(waves * 32, B), B, 0, stream>>>(
            x, W1t, deg, b1, hW1, agg1);
    }
    k_scatter128<<<blocks((long long)N_EDGES * 32, B), B, 0, stream>>>(
        hW1, src, dst, deg, agg1);
    k_relu<<<blocks((long long)N_NODES * HID_DIM, B), B, 0, stream>>>(
        agg1, (long long)N_NODES * HID_DIM);

    // layer 2: GEMM (+self-loop+bias epilogue direct to d_out) -> edge scatter
    {
        long long waves = (long long)(N_NODES / 16) * ((N_CLASSES + 15) / 16);
        k_gemm_bf16<N_CLASSES><<<blocks(waves * 32, B), B, 0, stream>>>(
            agg1, W2t, deg, b2, hW2, out);
    }
    k_scatter40<<<blocks((long long)N_EDGES * 32, B), B, 0, stream>>>(
        hW2, src, dst, deg, out);
}
